// MultiHeadAttention_10642928959690
// MI455X (gfx1250) — matrix operand order repaired and checked by reading, not compiled, not run on hardware
//
#include <hip/hip_runtime.h>
#include <hip/hip_bf16.h>

// MI455X (gfx1250) fused MHA block:
//   qkv proj (WMMA bf16, LDS-staged) -> causal scores (WMMA) -> row softmax
//   -> P@V (WMMA, V tiles streamed by the Tensor Data Mover into LDS with
//   TENSORcnt double-buffering) -> out proj + residual (WMMA) -> layernorm.
// The materialized attn matrix (1 GB, mandated by the reference's second
// output) dominates the roofline; all GEMM math runs on v_wmma_f32_16x16x32_bf16.

typedef __attribute__((ext_vector_type(16))) __bf16 v16bf;
typedef __attribute__((ext_vector_type(8)))  float  v8f;
typedef __attribute__((ext_vector_type(4)))  unsigned int v4u;
typedef __attribute__((ext_vector_type(8)))  int v8i;
typedef __attribute__((ext_vector_type(4)))  int v4i;

#define B_   4
#define S_   2048
#define H_   1024
#define NH_  16
#define D_   64
#define M_   (B_ * S_)          // 8192 token rows
#define SCALE_ 0.03125f         // 1/sqrt(H) = 1/32
#define EPS_   1e-5f
#define NEG_INF (-__builtin_inff())

#define HAVE_TDM __has_builtin(__builtin_amdgcn_tensor_load_to_lds)

static __device__ inline v8f wmma_bf16(v16bf a, v16bf b, v8f c) {
  // (neg_a, A, neg_b, B, c_mod, C, reuse_a, reuse_b)
  return __builtin_amdgcn_wmma_f32_16x16x32_bf16(false, a, false, b, (short)0, c,
                                                 false, false);
}

// ---- Tensor Data Mover: 2D bf16 tile (tile_d0 x tile_d1) -> LDS ---------
// D# packing per CDNA5 ISA ch.10.8 (group0: count/lds/global/type,
// group1: data_size, tensor dims, tile dims, stride; groups 2/3 zero for 2D).
static __device__ inline void tdm_load_2d_bf16(unsigned lds_addr,
                                               const __bf16* gptr,
                                               unsigned tensor_d0, unsigned tensor_d1,
                                               unsigned tile_d0, unsigned tile_d1,
                                               unsigned stride0) {
#if HAVE_TDM
  const unsigned long long ga = (unsigned long long)gptr;
  v4u g0;
  g0[0] = 1u;                                        // count=1 (valid user D#)
  g0[1] = lds_addr;                                  // LDS byte address [63:32]
  g0[2] = (unsigned)(ga & 0xFFFFFFFFu);              // global_addr[31:0]
  g0[3] = (unsigned)((ga >> 32) & 0x1FFFFFFu)        // global_addr[56:32]
        | (2u << 30);                                // type=2 ("image")
  v8i g1;
  g1[0] = (int)(1u << 16);                           // data_size=1 -> 2 bytes
  g1[1] = (int)(tensor_d0 << 16);                    // tensor_dim0[15:0]
  g1[2] = (int)((tensor_d0 >> 16) | (tensor_d1 << 16));
  g1[3] = (int)((tensor_d1 >> 16) | (tile_d0 << 16));
  g1[4] = (int)(tile_d1 & 0xFFFFu);                  // tile_dim1, tile_dim2=0
  g1[5] = (int)stride0;                              // tensor_dim0_stride[31:0]
  g1[6] = 0;
  g1[7] = 0;
  const v4i g2 = {0, 0, 0, 0};                       // 2D tensor: groups 2/3 unused
  const v4i g3 = {0, 0, 0, 0};
#if defined(__clang_major__) && (__clang_major__ >= 23)
  const v8i g4 = {0, 0, 0, 0, 0, 0, 0, 0};
  __builtin_amdgcn_tensor_load_to_lds(g0, g1, g2, g3, g4, 0);
#else
  __builtin_amdgcn_tensor_load_to_lds(g0, g1, g2, g3, 0);
#endif
#else
  (void)lds_addr; (void)gptr; (void)tensor_d0; (void)tensor_d1;
  (void)tile_d0; (void)tile_d1; (void)stride0;
#endif
}

// ---- fragment loaders matching ISA 7.12.2 wave32 layouts ----------------

// A fragment 16x32 (MxK), element (m,k) at base[m*stride + k]
static __device__ inline v16bf load_a_bf16(const __bf16* base, int stride, int lane) {
  const int m    = lane & 15;
  const int koff = (lane & 16) ? 8 : 0;   // lanes 16-31 hold K 8-15 / 24-31
  const __bf16* p = base + m * stride + koff;
  v16bf a;
#pragma unroll
  for (int e = 0; e < 8; ++e) { a[e] = p[e]; a[e + 8] = p[e + 16]; }
  return a;
}

// A fragment 16x32 from fp32 memory, converted to bf16 in-register
static __device__ inline v16bf load_a_f32(const float* base, long stride, int lane) {
  const int m    = lane & 15;
  const int koff = (lane & 16) ? 8 : 0;
  const float* p = base + (long)m * stride + koff;
  v16bf a;
#pragma unroll
  for (int e = 0; e < 8; ++e) { a[e] = (__bf16)p[e]; a[e + 8] = (__bf16)p[e + 16]; }
  return a;
}

// B fragment 32x16 (KxN), element (k,n) at base[n*stride + k]   (W[n][k] rows)
static __device__ inline v16bf load_b_rowmajor(const __bf16* base, int stride, int lane) {
  const int n  = lane & 15;
  const int k0 = (lane & 16) ? 8 : 0;     // lanes 16-31 hold K 8-15 / 24-31
  const __bf16* p = base + n * stride + k0;
  v16bf b;
#pragma unroll
  for (int e = 0; e < 8; ++e) { b[e] = p[e]; b[e + 8] = p[e + 16]; }
  return b;
}

// B fragment 32x16 (KxN), element (k,n) at base[k*stride + n]   (V[j][d] rows)
static __device__ inline v16bf load_b_colmajor(const __bf16* base, int stride, int lane) {
  const int n  = lane & 15;
  const int k0 = (lane & 16) ? 8 : 0;
  const __bf16* p = base + k0 * stride + n;
  v16bf b;
#pragma unroll
  for (int e = 0; e < 8; ++e) { b[e] = p[e * stride]; b[e + 8] = p[(e + 16) * stride]; }
  return b;
}

// ---- kernel 1: QKV projections, out = X @ W^T + b, head-split bf16 ------

__global__ __launch_bounds__(256)
void qkv_proj_kernel(const float* __restrict__ X, const float* __restrict__ W,
                     const float* __restrict__ bias, __bf16* __restrict__ out_heads) {
  __shared__ __bf16 Al[64 * 32];    // 64 rows  x Kc=32
  __shared__ __bf16 Bl[128 * 32];   // 128 cols x Kc=32
  const int tid  = threadIdx.x;
  const int lane = tid & 31;
  const int wave = tid >> 5;
  const int bn = blockIdx.x & 7;           // H_/128 = 8 N-tiles
  const int bm = blockIdx.x >> 3;          // M_/64  = 128 M-tiles
  const int m0 = bm * 64, n0 = bn * 128;
  const int wm = (wave >> 2) * 32;         // wave grid 2 (M) x 4 (N)
  const int wn = (wave & 3) * 32;
  v8f acc[2][2] = {};
  for (int k0 = 0; k0 < H_; k0 += 32) {
#pragma unroll
    for (int i = 0; i < 8; ++i) {          // stage A: 64x32 fp32 -> bf16
      int idx = tid * 8 + i;
      int r = idx >> 5, c = idx & 31;
      Al[idx] = (__bf16)X[(long)(m0 + r) * H_ + k0 + c];
    }
#pragma unroll
    for (int i = 0; i < 16; ++i) {         // stage B: 128x32 fp32 -> bf16
      int idx = tid * 16 + i;
      int r = idx >> 5, c = idx & 31;
      Bl[idx] = (__bf16)W[(long)(n0 + r) * H_ + k0 + c];
    }
    __syncthreads();
    if (k0 + 32 < H_) {                    // global_prefetch_b8 of next K tile
      __builtin_prefetch(&X[(long)(m0 + (tid >> 2)) * H_ + k0 + 32], 0, 1);
      __builtin_prefetch(&W[(long)(n0 + (tid >> 1)) * H_ + k0 + 32], 0, 1);
    }
    v16bf a0 = load_a_bf16(&Al[(wm + 0) * 32], 32, lane);
    v16bf a1 = load_a_bf16(&Al[(wm + 16) * 32], 32, lane);
    v16bf b0 = load_b_rowmajor(&Bl[(wn + 0) * 32], 32, lane);
    v16bf b1 = load_b_rowmajor(&Bl[(wn + 16) * 32], 32, lane);
    acc[0][0] = wmma_bf16(a0, b0, acc[0][0]);
    acc[0][1] = wmma_bf16(a0, b1, acc[0][1]);
    acc[1][0] = wmma_bf16(a1, b0, acc[1][0]);
    acc[1][1] = wmma_bf16(a1, b1, acc[1][1]);
    __syncthreads();
  }
  const int rbase = (lane & 16) ? 8 : 0;   // C/D: row = vgpr + 8*(lane>=16)
  const int nn = lane & 15;
#pragma unroll
  for (int mi = 0; mi < 2; ++mi)
#pragma unroll
    for (int ni = 0; ni < 2; ++ni)
#pragma unroll
      for (int r = 0; r < 8; ++r) {
        int m = m0 + wm + mi * 16 + rbase + r;
        int n = n0 + wn + ni * 16 + nn;
        float val = acc[mi][ni][r] + bias[n];
        int b = m >> 11, s = m & (S_ - 1);
        int h = n >> 6,  d = n & (D_ - 1);
        out_heads[(((long)b * NH_ + h) * S_ + s) * D_ + d] = (__bf16)val;
      }
}

// ---- kernel 2: causal scores = scale * Qh @ Kh^T, -inf mask -------------

__global__ __launch_bounds__(128)
void scores_kernel(const __bf16* __restrict__ Qh, const __bf16* __restrict__ Kh,
                   float* __restrict__ attn) {
  const int lane = threadIdx.x & 31;
  const int wave = threadIdx.x >> 5;
  const int idx  = blockIdx.x * 4 + wave;          // one wave = one 16x16 tile
  const int TT   = (S_ / 16) * (S_ / 16);          // 16384 tiles per head
  const int bh   = idx / TT;
  const int rem  = idx % TT;
  const int i0 = (rem >> 7) * 16;
  const int j0 = (rem & 127) * 16;
  float* orow = attn + ((long)bh * S_ + i0) * S_ + j0;
  const int rbase = (lane & 16) ? 8 : 0;
  const int nn = lane & 15;
  if (j0 > i0 + 15) {                              // fully masked tile: skip WMMA
#pragma unroll
    for (int r = 0; r < 8; ++r) orow[(long)(rbase + r) * S_ + nn] = NEG_INF;
    return;                                        // wave-uniform branch
  }
  const __bf16* qb = Qh + ((long)bh * S_ + i0) * D_;
  const __bf16* kb = Kh + ((long)bh * S_ + j0) * D_;
  v8f c = {};
#pragma unroll
  for (int dd = 0; dd < D_; dd += 32) {
    v16bf a = load_a_bf16(qb + dd, D_, lane);
    v16bf b = load_b_rowmajor(kb + dd, D_, lane);
    c = wmma_bf16(a, b, c);
  }
#pragma unroll
  for (int r = 0; r < 8; ++r) {
    int i = i0 + rbase + r, j = j0 + nn;
    float val = c[r] * SCALE_;
    if (j > i) val = NEG_INF;
    orow[(long)(rbase + r) * S_ + nn] = val;
  }
}

// ---- kernel 3: in-place row softmax over 2048 keys ----------------------

__global__ __launch_bounds__(256)
void softmax_kernel(float* __restrict__ attn) {
  __shared__ float red[256];
  const long row = blockIdx.x;
  float* p = attn + row * S_;
  const int t = threadIdx.x;
  float v[8];
  float mx = NEG_INF;
#pragma unroll
  for (int i = 0; i < 8; ++i) { v[i] = p[t + i * 256]; mx = fmaxf(mx, v[i]); }
  red[t] = mx; __syncthreads();
  for (int off = 128; off > 0; off >>= 1) {
    if (t < off) red[t] = fmaxf(red[t], red[t + off]);
    __syncthreads();
  }
  mx = red[0]; __syncthreads();
  float s = 0.f;
#pragma unroll
  for (int i = 0; i < 8; ++i) { v[i] = __expf(v[i] - mx); s += v[i]; }
  red[t] = s; __syncthreads();
  for (int off = 128; off > 0; off >>= 1) {
    if (t < off) red[t] += red[t + off];
    __syncthreads();
  }
  const float inv = 1.f / red[0];
#pragma unroll
  for (int i = 0; i < 8; ++i) p[t + i * 256] = v[i] * inv;
}

// ---- kernel 4: ctx = attn @ Vh --------------------------------------------
// V tiles (32 keys x 16 dims, bf16) are streamed into per-wave LDS buffers by
// the Tensor Data Mover; TENSORcnt double-buffering overlaps DMA with WMMA.
// K-loop stops at the causal diagonal (probs past it are exactly 0).

__global__ __launch_bounds__(128)
void attn_v_kernel(const float* __restrict__ attn, const __bf16* __restrict__ Vh,
                   __bf16* __restrict__ ctx) {
#if HAVE_TDM
  __shared__ __bf16 Vl[4][2][32 * 16];             // 4 waves x double buffer x 1KB
#endif
  const int lane = threadIdx.x & 31;
  const int wave = threadIdx.x >> 5;
  const int idx  = blockIdx.x * 4 + wave;
  const int TI = S_ / 16, TD = D_ / 16;            // 128 x 4 tiles per head
  const int bh  = idx / (TI * TD);
  const int rem = idx % (TI * TD);
  const int i0 = (rem / TD) * 16;
  const int d0 = (rem % TD) * 16;
  const float*  ab = attn + ((long)bh * S_ + i0) * S_;
  const __bf16* vb = Vh + (long)bh * S_ * D_ + d0; // tile column origin
  v8f c = {};
  const int nt = (i0 + 16 + 31) / 32;              // 32-key tiles up to diagonal
#if HAVE_TDM
  tdm_load_2d_bf16((unsigned)(unsigned long long)&Vl[wave][0][0],
                   vb, /*tensor_d0=*/D_, /*tensor_d1=*/S_,
                   /*tile_d0=*/16, /*tile_d1=*/32, /*stride0=*/D_);
  for (int t = 0; t < nt; ++t) {
    const int j0 = t * 32;
    const int cur = t & 1;
    if (t + 1 < nt) {                              // prefetch next tile via TDM
      tdm_load_2d_bf16((unsigned)(unsigned long long)&Vl[wave][cur ^ 1][0],
                       vb + (long)(j0 + 32) * D_, D_, S_ - (j0 + 32), 16, 32, D_);
      __builtin_amdgcn_s_wait_tensorcnt(1);        // tile t resident, t+1 in flight
    } else {
      __builtin_amdgcn_s_wait_tensorcnt(0);
    }
    asm volatile("" ::: "memory");                 // keep ds reads after the wait
    v16bf a = load_a_f32(ab + j0, S_, lane);       // probs fp32 -> bf16 A-frag
    v16bf b = load_b_colmajor(&Vl[wave][cur][0], 16, lane);
    c = wmma_bf16(a, b, c);
  }
#else
  for (int t = 0; t < nt; ++t) {
    const int j0 = t * 32;
    v16bf a = load_a_f32(ab + j0, S_, lane);
    v16bf b = load_b_colmajor(vb + (long)j0 * D_, D_, lane);
    c = wmma_bf16(a, b, c);
  }
#endif
  const int b_ = bh >> 4, h = bh & (NH_ - 1);
  const int rbase = (lane & 16) ? 8 : 0;
  const int nn = lane & 15;
#pragma unroll
  for (int r = 0; r < 8; ++r) {
    int i = i0 + rbase + r;
    ctx[((long)b_ * S_ + i) * H_ + h * D_ + d0 + nn] = (__bf16)c[r];
  }
}

// ---- kernel 5: out proj + bias + residual (pre-LN, fp32) ----------------

__global__ __launch_bounds__(256)
void out_proj_kernel(const __bf16* __restrict__ Xb, const float* __restrict__ W,
                     const float* __restrict__ bias, const float* __restrict__ residual,
                     float* __restrict__ pre) {
  __shared__ __bf16 Al[64 * 32];
  __shared__ __bf16 Bl[128 * 32];
  const int tid  = threadIdx.x;
  const int lane = tid & 31;
  const int wave = tid >> 5;
  const int bn = blockIdx.x & 7;
  const int bm = blockIdx.x >> 3;
  const int m0 = bm * 64, n0 = bn * 128;
  const int wm = (wave >> 2) * 32;
  const int wn = (wave & 3) * 32;
  v8f acc[2][2] = {};
  for (int k0 = 0; k0 < H_; k0 += 32) {
#pragma unroll
    for (int i = 0; i < 8; ++i) {
      int idx = tid * 8 + i;
      int r = idx >> 5, c = idx & 31;
      Al[idx] = Xb[(long)(m0 + r) * H_ + k0 + c];
    }
#pragma unroll
    for (int i = 0; i < 16; ++i) {
      int idx = tid * 16 + i;
      int r = idx >> 5, c = idx & 31;
      Bl[idx] = (__bf16)W[(long)(n0 + r) * H_ + k0 + c];
    }
    __syncthreads();
    if (k0 + 32 < H_) {
      __builtin_prefetch(&Xb[(long)(m0 + (tid >> 2)) * H_ + k0 + 32], 0, 1);
      __builtin_prefetch(&W[(long)(n0 + (tid >> 1)) * H_ + k0 + 32], 0, 1);
    }
    v16bf a0 = load_a_bf16(&Al[(wm + 0) * 32], 32, lane);
    v16bf a1 = load_a_bf16(&Al[(wm + 16) * 32], 32, lane);
    v16bf b0 = load_b_rowmajor(&Bl[(wn + 0) * 32], 32, lane);
    v16bf b1 = load_b_rowmajor(&Bl[(wn + 16) * 32], 32, lane);
    acc[0][0] = wmma_bf16(a0, b0, acc[0][0]);
    acc[0][1] = wmma_bf16(a0, b1, acc[0][1]);
    acc[1][0] = wmma_bf16(a1, b0, acc[1][0]);
    acc[1][1] = wmma_bf16(a1, b1, acc[1][1]);
    __syncthreads();
  }
  const int rbase = (lane & 16) ? 8 : 0;
  const int nn = lane & 15;
#pragma unroll
  for (int mi = 0; mi < 2; ++mi)
#pragma unroll
    for (int ni = 0; ni < 2; ++ni)
#pragma unroll
      for (int r = 0; r < 8; ++r) {
        int m = m0 + wm + mi * 16 + rbase + r;
        int n = n0 + wn + ni * 16 + nn;
        float val = acc[mi][ni][r] + bias[n] + residual[(long)m * H_ + n];
        pre[(long)m * H_ + n] = val;
      }
}

// ---- kernel 6: layernorm over H=1024 ------------------------------------

__global__ __launch_bounds__(256)
void layernorm_kernel(const float* __restrict__ x_, const float* __restrict__ gamma,
                      const float* __restrict__ beta, float* __restrict__ out) {
  __shared__ float red[256];
  const long row = blockIdx.x;
  const float* x = x_ + row * H_;
  float* o = out + row * H_;
  const int t = threadIdx.x;
  float v[4];
  float s = 0.f;
#pragma unroll
  for (int i = 0; i < 4; ++i) { v[i] = x[t + i * 256]; s += v[i]; }
  red[t] = s; __syncthreads();
  for (int off = 128; off > 0; off >>= 1) {
    if (t < off) red[t] += red[t + off];
    __syncthreads();
  }
  const float mu = red[0] * (1.f / H_);
  __syncthreads();
  float s2 = 0.f;
#pragma unroll
  for (int i = 0; i < 4; ++i) { float d = v[i] - mu; s2 += d * d; }
  red[t] = s2; __syncthreads();
  for (int off = 128; off > 0; off >>= 1) {
    if (t < off) red[t] += red[t + off];
    __syncthreads();
  }
  const float rs = rsqrtf(red[0] * (1.f / H_) + EPS_);
#pragma unroll
  for (int i = 0; i < 4; ++i) {
    int c = t + i * 256;
    o[c] = (v[i] - mu) * rs * gamma[c] + beta[c];
  }
}

// ---- host launch --------------------------------------------------------

extern "C" void kernel_launch(void* const* d_in, const int* in_sizes, int n_in,
                              void* d_out, int out_size, void* d_ws, size_t ws_size,
                              hipStream_t stream) {
  (void)in_sizes; (void)n_in; (void)out_size; (void)ws_size;
  const float* q  = (const float*)d_in[0];
  const float* k  = (const float*)d_in[1];
  const float* v  = (const float*)d_in[2];
  // d_in[3] is the causal mask; semantics reproduced from indices instead.
  const float* Wq = (const float*)d_in[4];
  const float* bq = (const float*)d_in[5];
  const float* Wk = (const float*)d_in[6];
  const float* bk = (const float*)d_in[7];
  const float* Wv = (const float*)d_in[8];
  const float* bv = (const float*)d_in[9];
  const float* Wm = (const float*)d_in[10];
  const float* bm = (const float*)d_in[11];
  const float* gamma = (const float*)d_in[12];
  const float* beta  = (const float*)d_in[13];

  float* out  = (float*)d_out;                     // [B,S,H]
  float* attn = out + (long)M_ * H_;               // [B,NH,S,S] lives in d_out

  const long proj_elems = (long)M_ * H_;           // 8M elements
  char* wsp = (char*)d_ws;
  __bf16* Qh  = (__bf16*)wsp; wsp += proj_elems * 2;   // [B,NH,S,D]
  __bf16* Kh  = (__bf16*)wsp; wsp += proj_elems * 2;
  __bf16* Vh  = (__bf16*)wsp; wsp += proj_elems * 2;
  __bf16* ctx = (__bf16*)wsp; wsp += proj_elems * 2;   // [B,S,H]
  float*  pre = (float*)wsp;                           // pre-LN [B,S,H]

  const int gemm_blocks = (M_ / 64) * (H_ / 128);      // 1024
  qkv_proj_kernel<<<gemm_blocks, 256, 0, stream>>>(q, Wq, bq, Qh);
  qkv_proj_kernel<<<gemm_blocks, 256, 0, stream>>>(k, Wk, bk, Kh);
  qkv_proj_kernel<<<gemm_blocks, 256, 0, stream>>>(v, Wv, bv, Vh);

  const int score_tiles = B_ * NH_ * (S_ / 16) * (S_ / 16);  // 1,048,576
  scores_kernel<<<score_tiles / 4, 128, 0, stream>>>(Qh, Kh, attn);

  softmax_kernel<<<B_ * NH_ * S_, 256, 0, stream>>>(attn);   // 131,072 rows

  const int pv_tiles = B_ * NH_ * (S_ / 16) * (D_ / 16);     // 32,768
  attn_v_kernel<<<pv_tiles / 4, 128, 0, stream>>>(attn, Vh, ctx);

  out_proj_kernel<<<gemm_blocks, 256, 0, stream>>>(ctx, Wm, bm, q, pre);

  layernorm_kernel<<<M_, 256, 0, stream>>>(pre, gamma, beta, out);
}
